// CopeMultiHeadAttention_30966714204560
// MI455X (gfx1250) — compile-verified, matrix operand order checked by reading, not run on hardware
//
#include <hip/hip_runtime.h>
#include <hip/hip_bf16.h>

// ---------------------------------------------------------------------------
// CoPE multi-head attention for MI455X (gfx1250), wave32, bf16 WMMA path.
// B=4, S=1024, D=1024, H=16, HD=64, NPOS=64.
// ---------------------------------------------------------------------------

#define B_  4
#define S_  1024
#define D_  1024
#define H_  16
#define HD_ 64
#define NPOS_ 64

typedef __attribute__((ext_vector_type(16))) __bf16 v16bf;
typedef __attribute__((ext_vector_type(8)))  float  v8f;

union BfVec {
    v16bf  v;
    __bf16 e[16];
};

// K-offset of element i of a 16-bit A/B WMMA fragment (16x16x32) for `lane`.
// ISA 7.12.2: lanes 0-15 hold K={0..7,16..23}, lanes 16-31 hold K={8..15,24..31}.
static __device__ __forceinline__ int kmap(int i, int lane) {
    int p = i >> 1, q = i & 1;
    return ((p & 3) << 1) + q + ((p >> 2) << 4) + ((lane >> 4) << 3);
}

static __device__ __forceinline__ v8f wmma_bf16(const BfVec& a, const BfVec& b, v8f c) {
    return __builtin_amdgcn_wmma_f32_16x16x32_bf16(false, a.v, false, b.v,
                                                   (short)0, c, false, false);
}

// ---------------------------------------------------------------------------
// Kernel 0: one-time fp32 -> bf16 precast (native v_cvt, packed by compiler).
// ---------------------------------------------------------------------------
__global__ void cope_cvt_bf16(const float* __restrict__ src,
                              __bf16* __restrict__ dst, int n) {
    int i = (blockIdx.x * blockDim.x + threadIdx.x) * 4;
    if (i + 3 < n) {
        float4 f = *(const float4*)(src + i);
        dst[i + 0] = (__bf16)f.x;
        dst[i + 1] = (__bf16)f.y;
        dst[i + 2] = (__bf16)f.z;
        dst[i + 3] = (__bf16)f.w;
    }
}

// ---------------------------------------------------------------------------
// Kernel 1: Q/K/V projections.  y = x @ W^T + b, bf16 out in [B,H,S,HD].
// One wave computes a 16(M) x 64(N) strip; K=1024 in 32 steps of 32.
// All operands pre-cast to bf16 -> inner loop is loads + WMMA only.
// ---------------------------------------------------------------------------
__global__ void cope_proj_qkv(const __bf16* __restrict__ Xq,
                              const __bf16* __restrict__ Xk,
                              const __bf16* __restrict__ Xv,
                              const __bf16* __restrict__ Wq, const float* __restrict__ bq,
                              const __bf16* __restrict__ Wk, const float* __restrict__ bk,
                              const __bf16* __restrict__ Wv, const float* __restrict__ bv,
                              __bf16* __restrict__ Qbf,
                              __bf16* __restrict__ Kbf,
                              __bf16* __restrict__ Vbf) {
    const int lane = threadIdx.x & 31;
    const int wave = threadIdx.x >> 5;
    const int strip = blockIdx.x * 4 + wave;           // 3 * 256 * 16 strips
    const int mat    = strip / 4096;                   // 0:q 1:k 2:v
    const int rem    = strip - mat * 4096;
    const int mtile  = rem & 255;                      // 256 row tiles (rows of 16)
    const int nstrip = rem >> 8;                       // 16 col strips (64 cols)

    const __bf16* X  = (mat == 0) ? Xq : (mat == 1) ? Xk : Xv;
    const __bf16* W  = (mat == 0) ? Wq : (mat == 1) ? Wk : Wv;
    const float*  bi = (mat == 0) ? bq : (mat == 1) ? bk : bv;
    __bf16* O = (mat == 0) ? Qbf : (mat == 1) ? Kbf : Vbf;

    const int row0 = mtile * 16;
    const int col0 = nstrip * 64;
    const int m = lane & 15;
    const int n = lane & 15;

    v8f acc[4] = {};
    for (int k0 = 0; k0 < D_; k0 += 32) {
        if (k0 + 32 < D_)   // prefetch next A slab (global_prefetch_b8)
            __builtin_prefetch(X + (size_t)(row0 + m) * D_ + k0 + 32, 0, 3);
        BfVec a;
#pragma unroll
        for (int i = 0; i < 16; ++i)
            a.e[i] = X[(size_t)(row0 + m) * D_ + k0 + kmap(i, lane)];
#pragma unroll
        for (int t = 0; t < 4; ++t) {
            BfVec bm;
            const int o = col0 + t * 16 + n;           // B[k][n] = W[o][d]
#pragma unroll
            for (int i = 0; i < 16; ++i)
                bm.e[i] = W[(size_t)o * D_ + k0 + kmap(i, lane)];
            acc[t] = wmma_bf16(a, bm, acc[t]);
        }
    }
    // store bf16 in [B,H,S,HD]
#pragma unroll
    for (int t = 0; t < 4; ++t) {
        const int o  = col0 + t * 16 + n;
        const float bias = bi[o];
        const int h  = o >> 6;
        const int hd = o & 63;
#pragma unroll
        for (int r = 0; r < 8; ++r) {
            const int M  = r + ((lane >> 4) << 3);
            const int rr = row0 + M;
            const int b  = rr >> 10;
            const int s  = rr & 1023;
            O[(((size_t)(b * H_ + h) * S_) + s) * HD_ + hd] = (__bf16)(acc[t][r] + bias);
        }
    }
}

// ---------------------------------------------------------------------------
// Kernel 2: fused CoPE attention per (b, h, 16-query tile). 128 threads.
//   phase1 : logits = (Q K^T)/8 -> LDS bf16 [16][1024]; logits_int -> LDS f32
//   phase2 : sigmoid gates; reverse suffix-scan; CoPE interp; softmax
//   phase3 : out = P @ V via WMMA, bf16 out in [B,S,D]
// ---------------------------------------------------------------------------
__global__ void cope_attn(const __bf16* __restrict__ Qbf,
                          const __bf16* __restrict__ Kbf,
                          const __bf16* __restrict__ Vbf,
                          const __bf16* __restrict__ peb,   // [HD][NPOS] bf16
                          __bf16* __restrict__ Abf) {
    __shared__ __bf16 lg[16 * S_];              // 32 KB: raw logits, then probs
    __shared__ float li[16 * NPOS_];            // 4 KB : logits_int
    __shared__ float scratch[2 * 64];           // segment totals / reductions
    __shared__ float rowmax[16];
    __shared__ float rowsum[16];

    const int lane = threadIdx.x & 31;
    const int wave = threadIdx.x >> 5;          // 4 waves
    const int qt = blockIdx.x & 63;             // 64 q-tiles
    const int bh = blockIdx.x >> 6;             // b*H + h
    const int b  = bh >> 4;
    const int h  = bh & 15;
    const int q0 = qt * 16;
    const size_t hb = (size_t)bh * S_ * HD_;    // head base in Q/K/V bf16

    const int m = lane & 15;
    const int n = lane & 15;

    // --- phase 1: A fragments of the Q tile (K = 64, two 32-wide steps) ---
    BfVec a0, a1;
#pragma unroll
    for (int i = 0; i < 16; ++i) {
        const int kk = kmap(i, lane);
        a0.e[i] = Qbf[hb + (size_t)(q0 + m) * HD_ + kk];
        a1.e[i] = Qbf[hb + (size_t)(q0 + m) * HD_ + 32 + kk];
    }

    // Q K^T: each wave covers 16 key-tiles (256 keys)
    for (int nt = wave * 16; nt < wave * 16 + 16; ++nt) {
        v8f c = {};
        BfVec b0, b1;
        const int kidx = nt * 16 + n;           // key index (column)
#pragma unroll
        for (int i = 0; i < 16; ++i) {
            const int kk = kmap(i, lane);
            b0.e[i] = Kbf[hb + (size_t)kidx * HD_ + kk];
            b1.e[i] = Kbf[hb + (size_t)kidx * HD_ + 32 + kk];
        }
        c = wmma_bf16(a0, b0, c);
        c = wmma_bf16(a1, b1, c);
#pragma unroll
        for (int r = 0; r < 8; ++r) {
            const int M = r + ((lane >> 4) << 3);
            lg[M * S_ + nt * 16 + n] = (__bf16)(c[r] * 0.125f);   // 1/sqrt(64)
        }
    }

    // logits_int = Q @ pos_emb  (pos_emb [HD, NPOS] row-major, unscaled)
    {
        v8f c = {};
        BfVec b0, b1;
        const int np = wave * 16 + n;
#pragma unroll
        for (int i = 0; i < 16; ++i) {
            const int kk = kmap(i, lane);
            b0.e[i] = peb[(size_t)kk * NPOS_ + np];
            b1.e[i] = peb[(size_t)(32 + kk) * NPOS_ + np];
        }
        c = wmma_bf16(a0, b0, c);
        c = wmma_bf16(a1, b1, c);
#pragma unroll
        for (int r = 0; r < 8; ++r) {
            const int M = r + ((lane >> 4) << 3);
            li[M * NPOS_ + wave * 16 + n] = c[r];
        }
    }
    __syncthreads();

    // --- phase 2: reverse cumsum of gates + CoPE + softmax (fp32) ---
    // 128 threads: 2 rows per pass, 64 segments of 16 keys per row.
    const int rloc = threadIdx.x >> 6;          // 0..1
    const int seg  = threadIdx.x & 63;          // 0..63
    for (int pass = 0; pass < 8; ++pass) {
        const int row  = pass * 2 + rloc;
        const int base = row * S_ + seg * 16;
        float raw[16], sfx[16];
        float run = 0.f;
#pragma unroll
        for (int i = 15; i >= 0; --i) {
            raw[i] = (float)lg[base + i];
            run += 1.f / (1.f + __expf(-raw[i]));  // sigmoid gate
            sfx[i] = run;
        }
        scratch[rloc * 64 + seg] = run;
        __syncthreads();
        float off = 0.f;
        for (int t = seg + 1; t < 64; ++t) off += scratch[rloc * 64 + t];
        __syncthreads();

        float mx = -1e30f;
#pragma unroll
        for (int i = 0; i < 16; ++i) {
            float pos = fminf(sfx[i] + off, (float)(NPOS_ - 1));
            float pf  = floorf(pos);
            float pc  = ceilf(pos);
            float w   = pos - pf;
            float lf  = li[row * NPOS_ + (int)pf];
            float lc  = li[row * NPOS_ + (int)pc];
            raw[i] = raw[i] + lc * w + lf * (1.f - w);
            mx = fmaxf(mx, raw[i]);
        }
        scratch[rloc * 64 + seg] = mx;
        __syncthreads();
        if (seg == 0) {
            float m2 = -1e30f;
            for (int t = 0; t < 64; ++t) m2 = fmaxf(m2, scratch[rloc * 64 + t]);
            rowmax[row] = m2;
        }
        __syncthreads();
        const float rm = rowmax[row];
        float ls = 0.f;
#pragma unroll
        for (int i = 0; i < 16; ++i) {
            raw[i] = __expf(raw[i] - rm);
            ls += raw[i];
        }
        scratch[rloc * 64 + seg] = ls;
        __syncthreads();
        if (seg == 0) {
            float s2 = 0.f;
            for (int t = 0; t < 64; ++t) s2 += scratch[rloc * 64 + t];
            rowsum[row] = s2;
        }
        __syncthreads();
        const float inv = 1.f / rowsum[row];
#pragma unroll
        for (int i = 0; i < 16; ++i)
            lg[base + i] = (__bf16)(raw[i] * inv);   // probs overwrite logits
        __syncthreads();
    }

    // --- phase 3: out = P @ V.  Each wave owns one 16-wide hd column tile ---
    {
        v8f c = {};
        const int hd = wave * 16 + n;
        for (int k0 = 0; k0 < S_; k0 += 32) {
            BfVec a, bm;
#pragma unroll
            for (int i = 0; i < 16; ++i) {
                const int kk = kmap(i, lane);
                a.e[i]  = lg[m * S_ + k0 + kk];
                bm.e[i] = Vbf[hb + (size_t)(k0 + kk) * HD_ + hd];
            }
            c = wmma_bf16(a, bm, c);
        }
        // store bf16 in [B, S, D] with col = h*64 + hd
#pragma unroll
        for (int r = 0; r < 8; ++r) {
            const int M = r + ((lane >> 4) << 3);
            Abf[((size_t)(b * S_ + q0 + M)) * D_ + h * HD_ + wave * 16 + n] = (__bf16)c[r];
        }
    }
}

// ---------------------------------------------------------------------------
// Kernel 3: output projection  y = attn @ Wo^T + bo  -> fp32 d_out [B,S,D]
// ---------------------------------------------------------------------------
__global__ void cope_proj_out(const __bf16* __restrict__ Abf,
                              const __bf16* __restrict__ Wo,
                              const float* __restrict__ bo,
                              float* __restrict__ out) {
    const int lane = threadIdx.x & 31;
    const int wave = threadIdx.x >> 5;
    const int strip = blockIdx.x * 4 + wave;   // 256 * 16 strips
    const int mtile  = strip & 255;
    const int nstrip = strip >> 8;
    const int row0 = mtile * 16;
    const int col0 = nstrip * 64;
    const int m = lane & 15;
    const int n = lane & 15;

    v8f acc[4] = {};
    for (int k0 = 0; k0 < D_; k0 += 32) {
        if (k0 + 32 < D_)
            __builtin_prefetch(Abf + (size_t)(row0 + m) * D_ + k0 + 32, 0, 3);
        BfVec a;
#pragma unroll
        for (int i = 0; i < 16; ++i)
            a.e[i] = Abf[(size_t)(row0 + m) * D_ + k0 + kmap(i, lane)];
#pragma unroll
        for (int t = 0; t < 4; ++t) {
            BfVec bm;
            const int o = col0 + t * 16 + n;
#pragma unroll
            for (int i = 0; i < 16; ++i)
                bm.e[i] = Wo[(size_t)o * D_ + k0 + kmap(i, lane)];
            acc[t] = wmma_bf16(a, bm, acc[t]);
        }
    }
#pragma unroll
    for (int t = 0; t < 4; ++t) {
        const int o = col0 + t * 16 + n;
        const float bias = bo[o];
#pragma unroll
        for (int r = 0; r < 8; ++r) {
            const int M = r + ((lane >> 4) << 3);
            out[(size_t)(row0 + M) * D_ + o] = acc[t][r] + bias;
        }
    }
}

// ---------------------------------------------------------------------------
extern "C" void kernel_launch(void* const* d_in, const int* in_sizes, int n_in,
                              void* d_out, int out_size, void* d_ws, size_t ws_size,
                              hipStream_t stream) {
    const float* q   = (const float*)d_in[0];
    const float* k   = (const float*)d_in[1];
    const float* v   = (const float*)d_in[2];
    const float* Wq  = (const float*)d_in[3];
    const float* bq  = (const float*)d_in[4];
    const float* Wk  = (const float*)d_in[5];
    const float* bk  = (const float*)d_in[6];
    const float* Wv  = (const float*)d_in[7];
    const float* bv  = (const float*)d_in[8];
    const float* Wo  = (const float*)d_in[9];
    const float* bo  = (const float*)d_in[10];
    const float* pe  = (const float*)d_in[11];

    const size_t nAct = (size_t)B_ * S_ * D_;          // 4M elements
    const size_t nW   = (size_t)D_ * D_;               // 1M elements
    __bf16* p = (__bf16*)d_ws;
    __bf16* Qbf = p; p += nAct;
    __bf16* Kbf = p; p += nAct;
    __bf16* Vbf = p; p += nAct;
    __bf16* Abf = p; p += nAct;
    __bf16* Xq  = p; p += nAct;
    __bf16* Xk  = p; p += nAct;
    __bf16* Xv  = p; p += nAct;
    __bf16* Wqb = p; p += nW;
    __bf16* Wkb = p; p += nW;
    __bf16* Wvb = p; p += nW;
    __bf16* Wob = p; p += nW;
    __bf16* peb = p; p += HD_ * NPOS_;                 // ~64.5 MB total

    // 0) one-time fp32 -> bf16 precasts (activations, weights, pos_emb)
    cope_cvt_bf16<<<(int)(nAct / 1024), 256, 0, stream>>>(q,  Xq,  (int)nAct);
    cope_cvt_bf16<<<(int)(nAct / 1024), 256, 0, stream>>>(k,  Xk,  (int)nAct);
    cope_cvt_bf16<<<(int)(nAct / 1024), 256, 0, stream>>>(v,  Xv,  (int)nAct);
    cope_cvt_bf16<<<(int)(nW   / 1024), 256, 0, stream>>>(Wq, Wqb, (int)nW);
    cope_cvt_bf16<<<(int)(nW   / 1024), 256, 0, stream>>>(Wk, Wkb, (int)nW);
    cope_cvt_bf16<<<(int)(nW   / 1024), 256, 0, stream>>>(Wv, Wvb, (int)nW);
    cope_cvt_bf16<<<(int)(nW   / 1024), 256, 0, stream>>>(Wo, Wob, (int)nW);
    cope_cvt_bf16<<<4, 256, 0, stream>>>(pe, peb, HD_ * NPOS_);

    // 1) QKV projections: 3*256*16 strips, 4 waves/block of 128 threads
    cope_proj_qkv<<<3072, 128, 0, stream>>>(Xq, Xk, Xv, Wqb, bq, Wkb, bk, Wvb, bv,
                                            Qbf, Kbf, Vbf);
    // 2) fused CoPE attention: B*H*(S/16) = 4096 blocks of 128 threads
    cope_attn<<<4096, 128, 0, stream>>>(Qbf, Kbf, Vbf, peb, Abf);
    // 3) output projection: 256*16 strips / 4 waves = 1024 blocks
    cope_proj_out<<<1024, 128, 0, stream>>>(Abf, Wob, bo, (float*)d_out);
}
